// AttentionLayer_76570676953469
// MI455X (gfx1250) — matrix operand order repaired and checked by reading, not compiled, not run on hardware
//
#include <hip/hip_runtime.h>
#include <hip/hip_bf16.h>

typedef _Float16 half_t;
typedef __attribute__((ext_vector_type(16))) _Float16 v16h;
typedef __attribute__((ext_vector_type(8)))  _Float16 v8h;
typedef __attribute__((ext_vector_type(8)))  float    v8f;
typedef __attribute__((ext_vector_type(4)))  float    v4f;

#define BATCH 4096
#define SEQ 200
#define SEQ_PAD 208          // 13 tiles of 16
#define NTILES 13
#define EMB 128
#define HID 64
#define NWAVES 4
#define BLOCK_T (NWAVES * 32)

// ---------------- workspace layout ----------------
// [0, 32768)      : Wc   f32 [128][64]   (W1a + W1c, feeds per-batch cand projection)
// [32768, 49152)  : Bh   f16 packed WMMA-B layout [kc=4][nt=4][lane=32][j=16]
// [49152, 65536)  : Bp   f16 packed WMMA-B layout
#define WS_WC_BYTES   32768
#define WS_B_BYTES    16384

// ---------------- shared memory layout (din kernel) ----------------
// sHistF f32 [200][128] at dynamic-LDS offset 0 (async-copy target),
// then sCand f16 [128], then floats: sCp[64], sW2[64], sScores[208], sRed[128]
#define SM_HIST_FLOATS (SEQ * EMB)             // 25600
#define SM_CAND_HALFS  EMB                     // 128
#define SM_FLOATS      (64 + 64 + SEQ_PAD + BLOCK_T)
#define SMEM_BYTES (SM_HIST_FLOATS * 4 + SM_CAND_HALFS * 2 + SM_FLOATS * 4)

// Fold W1 (512x64) into Wc (f32) + Bh/Bp (f16, pre-swizzled into the WMMA
// B-operand layout: lane = (k%32)/8%2*16 + n%16, half j = k%8 + (k%32)/16*8).
__global__ __launch_bounds__(256) void prep_kernel(const float* __restrict__ W1,
                                                   float* __restrict__ Wc,
                                                   half_t* __restrict__ Bh,
                                                   half_t* __restrict__ Bp) {
    int idx = blockIdx.x * 256 + threadIdx.x;
    if (idx >= EMB * HID) return;
    int k = idx >> 6;
    int n = idx & 63;
    float wa = W1[(0 * EMB + k) * HID + n];   // cand block
    float wb = W1[(1 * EMB + k) * HID + n];   // hist block
    float wc = W1[(2 * EMB + k) * HID + n];   // (cand - hist) block
    float wd = W1[(3 * EMB + k) * HID + n];   // (cand * hist) block
    Wc[k * HID + n] = wa + wc;
    half_t wh = (half_t)(wb - wc);
    half_t wp = (half_t)wd;
    int kc   = k >> 5;          // 0..3  (K chunk of 32)
    int kin  = k & 31;
    int nt   = n >> 4;          // 0..3  (N tile of 16)
    int lane = (((kin >> 3) & 1) << 4) | (n & 15);
    int j    = (kin & 7) | ((kin >> 4) << 3);
    int off  = (((kc * 4 + nt) * 32 + lane) << 4) + j;
    Bh[off] = wh;
    Bp[off] = wp;
}

__global__ __launch_bounds__(BLOCK_T) void din_kernel(
    const float* __restrict__ cand,
    const float* __restrict__ hist,
    const int*   __restrict__ hmask,
    const float* __restrict__ b1,
    const float* __restrict__ W2,
    const float* __restrict__ Wc,
    const half_t* __restrict__ Bh,
    const half_t* __restrict__ Bp,
    float* __restrict__ out) {

    extern __shared__ char smem[];
    float*  sHistF  = (float*)smem;                     // offset 0 in dynamic LDS
    half_t* sCand   = (half_t*)(sHistF + SM_HIST_FLOATS);
    float*  sCp     = (float*)(sCand + SM_CAND_HALFS);
    float*  sW2     = sCp + 64;
    float*  sScores = sW2 + 64;
    float*  sRed    = sScores + SEQ_PAD;

    const int tid    = threadIdx.x;
    const int lane   = tid & 31;
    const int waveId = tid >> 5;
    const int mrow   = lane & 15;   // N-column inside a 16-wide tile / A row
    const int grp    = lane >> 4;   // half-wave group
    const int b      = blockIdx.x;

    const float* candb = cand + (size_t)b * EMB;
    const float* histb = hist + (size_t)b * SEQ * EMB;

    // ---- Phase 1 (issued first): async-copy hist[b] HBM -> LDS, no VGPR
    // round trip; 6400 x b128 chunks tracked on ASYNCcnt while we overlap
    // the per-batch projection below. sHistF sits at dynamic-LDS offset 0.
    for (int idx = tid; idx < SEQ * EMB / 4; idx += BLOCK_T) {
        unsigned     lds_off = (unsigned)(idx * 16);
        const float* gp      = histb + idx * 4;
        asm volatile("global_load_async_to_lds_b128 %0, %1, off"
                     :: "v"(lds_off), "v"(gp)
                     : "memory");
    }

    // ---- Phase 0 (overlapped with async copies): cand, W2, cp = b1 + cand@Wc
    sCand[tid] = (half_t)candb[tid];
    if (tid < 64) {
        sW2[tid] = W2[tid];
        float acc = b1[tid];
        #pragma unroll 4
        for (int k = 0; k < EMB; ++k)
            acc += candb[k] * Wc[k * HID + tid];
        sCp[tid] = acc;
    }
    if (tid < 8) sScores[SEQ + tid] = -1.0e9f;  // softmax pads (never rewritten)

#if __has_builtin(__builtin_amdgcn_s_wait_asynccnt)
    __builtin_amdgcn_s_wait_asynccnt(0);
#else
    asm volatile("s_wait_asynccnt 0x0" ::: "memory");
#endif
    __syncthreads();

    // ---- Phase 2: per 16-row seq tile, WMMA MLP + score reduction ----
    for (int t = waveId; t < NTILES; t += NWAVES) {
        // Build f16 A operands from f32 LDS hist.
        // 16x32 f16 A layout: lane<16 holds K {kb..kb+7, kb+16..kb+23}.
        // Tile 12 rows 200..207 read stale/OOB LDS: affects only score rows
        // >= 200, which are preset to -1e9 and never written (row-independent).
        v16h Ah[4], Ap[4];
        #pragma unroll
        for (int kc = 0; kc < 4; ++kc) {
            int kb = kc * 32 + grp * 8;
            const float* rp = &sHistF[(t * 16 + mrow) * EMB + kb];
            v4f f0 = *(const v4f*)(rp + 0);
            v4f f1 = *(const v4f*)(rp + 4);
            v4f f2 = *(const v4f*)(rp + 16);
            v4f f3 = *(const v4f*)(rp + 20);
            v16h ah = { (half_t)f0[0], (half_t)f0[1], (half_t)f0[2], (half_t)f0[3],
                        (half_t)f1[0], (half_t)f1[1], (half_t)f1[2], (half_t)f1[3],
                        (half_t)f2[0], (half_t)f2[1], (half_t)f2[2], (half_t)f2[3],
                        (half_t)f3[0], (half_t)f3[1], (half_t)f3[2], (half_t)f3[3] };
            Ah[kc] = ah;
            v8h clo = *(const v8h*)&sCand[kb];
            v8h chi = *(const v8h*)&sCand[kb + 16];
            v16h cc = __builtin_shufflevector(clo, chi,
                        0,1,2,3,4,5,6,7,8,9,10,11,12,13,14,15);
            Ap[kc] = ah * cc;   // (cand * hist) features, v_pk_mul_f16
        }

        float spart[8];
        #pragma unroll
        for (int i = 0; i < 8; ++i) spart[i] = 0.0f;

        #pragma unroll
        for (int nt = 0; nt < 4; ++nt) {
            float cpv = sCp[nt * 16 + mrow];  // b1 + cand-projection, folded in
            v8f acc;
            #pragma unroll
            for (int i = 0; i < 8; ++i) acc[i] = cpv;

            #pragma unroll
            for (int kc = 0; kc < 4; ++kc) {
                int boff = ((kc * 4 + nt) * 32 + lane) << 4;
                v16h bh = *(const v16h*)(Bh + boff);
                acc = __builtin_amdgcn_wmma_f32_16x16x32_f16(
                        false, Ah[kc], false, bh, (short)0, acc, false, false);
                v16h bp = *(const v16h*)(Bp + boff);
                acc = __builtin_amdgcn_wmma_f32_16x16x32_f16(
                        false, Ap[kc], false, bp, (short)0, acc, false, false);
            }
            float w2v = sW2[nt * 16 + mrow];
            #pragma unroll
            for (int i = 0; i < 8; ++i)
                spart[i] += fmaxf(acc[i], 0.0f) * w2v;   // relu(h) . W2 partials
        }

        // reduce across the 16 lanes holding the N-columns of each row
        #pragma unroll
        for (int i = 0; i < 8; ++i) {
            float x = spart[i];
            x += __shfl_xor(x, 1, 16);
            x += __shfl_xor(x, 2, 16);
            x += __shfl_xor(x, 4, 16);
            x += __shfl_xor(x, 8, 16);
            if (mrow == 0) {
                int s = t * 16 + i + grp * 8;   // lanes 0..15: M=i, 16..31: M=i+8
                if (s < SEQ)
                    sScores[s] = (hmask[(size_t)b * SEQ + s] == 0) ? -1.0e9f : x;
            }
        }
    }
    __syncthreads();

    // ---- Phase 3: block softmax over 208 scores (b2 drops out as a shift) ----
    float m = -3.0e38f;
    for (int s = tid; s < SEQ_PAD; s += BLOCK_T) m = fmaxf(m, sScores[s]);
    sRed[tid] = m;
    __syncthreads();
    for (int off = BLOCK_T / 2; off > 0; off >>= 1) {
        if (tid < off) sRed[tid] = fmaxf(sRed[tid], sRed[tid + off]);
        __syncthreads();
    }
    float mx = sRed[0];
    __syncthreads();

    float lsum = 0.0f;
    for (int s = tid; s < SEQ_PAD; s += BLOCK_T) {
        float e = __expf(sScores[s] - mx);
        sScores[s] = e;
        lsum += e;
    }
    sRed[tid] = lsum;
    __syncthreads();
    for (int off = BLOCK_T / 2; off > 0; off >>= 1) {
        if (tid < off) sRed[tid] += sRed[tid + off];
        __syncthreads();
    }
    float inv = 1.0f / sRed[0];
    __syncthreads();

    float* outAttn = out + (size_t)BATCH * EMB;
    for (int s = tid; s < SEQ_PAD; s += BLOCK_T) {
        float w = sScores[s] * inv;
        sScores[s] = w;
        if (s < SEQ) outAttn[(size_t)b * SEQ + s] = w;
    }
    __syncthreads();

    // ---- Phase 4: user_interest = attn @ hist, from LDS-resident f32 hist ----
    float uacc = 0.0f;
    #pragma unroll 4
    for (int s = 0; s < SEQ; ++s)
        uacc += sHistF[s * EMB + tid] * sScores[s];
    out[(size_t)b * EMB + tid] = uacc;
}

extern "C" void kernel_launch(void* const* d_in, const int* in_sizes, int n_in,
                              void* d_out, int out_size, void* d_ws, size_t ws_size,
                              hipStream_t stream) {
    const float* cand  = (const float*)d_in[0];
    const float* hist  = (const float*)d_in[1];
    const int*   mask  = (const int*)d_in[2];
    const float* W1    = (const float*)d_in[3];
    const float* b1    = (const float*)d_in[4];
    const float* W2    = (const float*)d_in[5];
    // d_in[6] (b2) is a uniform shift under softmax -> unused.
    float* out = (float*)d_out;

    float*  Wc = (float*)d_ws;
    half_t* Bh = (half_t*)((char*)d_ws + WS_WC_BYTES);
    half_t* Bp = (half_t*)((char*)d_ws + WS_WC_BYTES + WS_B_BYTES);

    // 104.5 KB dynamic LDS (> 64 KB default cap); gfx1250 WGP has 320 KB.
    (void)hipFuncSetAttribute((const void*)din_kernel,
                              hipFuncAttributeMaxDynamicSharedMemorySize,
                              (int)SMEM_BYTES);

    prep_kernel<<<(EMB * HID + 255) / 256, 256, 0, stream>>>(W1, Wc, Bh, Bp);
    din_kernel<<<BATCH, BLOCK_T, SMEM_BYTES, stream>>>(
        cand, hist, mask, b1, W2, Wc, Bh, Bp, out);
}
